// Attention_64905545777733
// MI455X (gfx1250) — compile-verified, matrix operand order checked by reading, not run on hardware
//
#include <hip/hip_runtime.h>
#include <hip/hip_bf16.h>

// Problem constants (reference: B=16, N=1024, D=768, H=12, hd=64)
#define B_   16
#define N_   1024
#define D_   768
#define H_   12
#define HD_  64
#define TD_  2304          // 3*D
#define BN_  (B_*N_)       // 16384 tokens

typedef __bf16 v4bf  __attribute__((ext_vector_type(4)));
typedef __bf16 v8bf  __attribute__((ext_vector_type(8)));
typedef __bf16 v16bf __attribute__((ext_vector_type(16)));
typedef float  v8f   __attribute__((ext_vector_type(8)));

#define WMMA_BF16(A, Bv, C) \
  __builtin_amdgcn_wmma_f32_16x16x32_bf16(false, (A), false, (Bv), (short)0, (C), false, false)

// Build a 16x32 bf16 WMMA operand for this lane from two contiguous 16B chunks.
// CDNA5 16-bit operand layout: lanes 0-15 hold K={kb..kb+7, kb+16..kb+23},
// lanes 16-31 hold K={kb+8..kb+15, kb+24..kb+31}  ->  p0 = row+kb+8g, p1 = row+kb+16+8g.
static __device__ __forceinline__ v16bf load_ab(const __bf16* p0, const __bf16* p1) {
  v8bf lo = *(const v8bf*)p0;
  v8bf hi = *(const v8bf*)p1;
  v16bf r;
#pragma unroll
  for (int i = 0; i < 8; ++i) { r[i] = lo[i]; r[i + 8] = hi[i]; }
  return r;
}

static __device__ __forceinline__ v8f zero8() {
  v8f z;
#pragma unroll
  for (int i = 0; i < 8; ++i) z[i] = 0.f;
  return z;
}

static __device__ __forceinline__ unsigned pk2(float a, float b) {
  union { __bf16 h[2]; unsigned u; } x;
  x.h[0] = (__bf16)a; x.h[1] = (__bf16)b;
  return x.u;
}

// ---------------- precision conversion kernels ----------------

__global__ void k_cvt_x(const float* __restrict__ x, __bf16* __restrict__ xb, int n4) {
  int i = blockIdx.x * blockDim.x + threadIdx.x;
  if (i >= n4) return;
  float4 v = ((const float4*)x)[i];
  v4bf o; o[0] = (__bf16)v.x; o[1] = (__bf16)v.y; o[2] = (__bf16)v.z; o[3] = (__bf16)v.w;
  ((v4bf*)xb)[i] = o;
}

// out[n*K + k] = bf16(in[k*Ncol + n])   (W [K,Ncol] -> W^T [Ncol,K], coalesced writes)
__global__ void k_cvt_wT(const float* __restrict__ w, __bf16* __restrict__ wT, int K, int Ncol) {
  int i = blockIdx.x * blockDim.x + threadIdx.x;
  if (i >= K * Ncol) return;
  int n = i / K, k = i - n * K;
  wT[i] = (__bf16)w[(size_t)k * Ncol + n];
}

// ---------------- QKV GEMM: qkv = x @ W_qkv, scattered into Q/K/Vt (bf16) ----------------
// One wave computes a 32x64 output tile (8 accumulators). K-step operands are
// double-buffered: loads for step k+1 are issued before the 8 WMMAs of step k.
// Q is pre-scaled by hd^-0.5 here so attention needs no per-logit multiply.

__global__ void __launch_bounds__(128) k_gemm_qkv(const __bf16* __restrict__ xb,
                                                  const __bf16* __restrict__ wT,
                                                  __bf16* __restrict__ Qh,
                                                  __bf16* __restrict__ Kh,
                                                  __bf16* __restrict__ Vt) {
  const int wid  = blockIdx.x * (blockDim.x >> 5) + (threadIdx.x >> 5);
  const int lane = threadIdx.x & 31;
  const int g = lane >> 4, r = lane & 15;
  const int NT = TD_ / 64;                 // 36 column tiles
  const int mt = wid / NT, nt = wid - mt * NT;
  const int row0 = mt * 32, col0 = nt * 64;

  const __bf16* arow0 = xb + (size_t)(row0 + r) * D_;
  const __bf16* arow1 = xb + (size_t)(row0 + 16 + r) * D_;
  const __bf16* bcol[4];
#pragma unroll
  for (int t = 0; t < 4; ++t) bcol[t] = wT + (size_t)(col0 + 16 * t + r) * D_;

  v8f acc[2][4];
#pragma unroll
  for (int m = 0; m < 2; ++m)
#pragma unroll
    for (int t = 0; t < 4; ++t) acc[m][t] = zero8();

  v16bf Ab[2][2], Bb[2][4];
  auto load_step = [&](int buf, int kb) {
    Ab[buf][0] = load_ab(arow0 + kb + 8 * g, arow0 + kb + 16 + 8 * g);
    Ab[buf][1] = load_ab(arow1 + kb + 8 * g, arow1 + kb + 16 + 8 * g);
#pragma unroll
    for (int t = 0; t < 4; ++t)
      Bb[buf][t] = load_ab(bcol[t] + kb + 8 * g, bcol[t] + kb + 16 + 8 * g);
  };

  load_step(0, 0);
  for (int kb = 0; kb < D_; kb += 32) {
    const int cur = (kb >> 5) & 1;
    if (kb + 32 < D_) load_step(cur ^ 1, kb + 32);   // prefetch next k-step
#pragma unroll
    for (int t = 0; t < 4; ++t) {
      acc[0][t] = WMMA_BF16(Ab[cur][0], Bb[cur][t], acc[0][t]);
      acc[1][t] = WMMA_BF16(Ab[cur][1], Bb[cur][t], acc[1][t]);
    }
  }

  // Epilogue: C layout -> lane holds col n = col0+16t+r, tokens m = j+8g in VGPR j.
  const int bb = row0 / N_;                 // 32-row tile never crosses a batch
  const int n0 = row0 - bb * N_;
#pragma unroll
  for (int m = 0; m < 2; ++m) {
    const int tok0 = n0 + 16 * m + 8 * g;
#pragma unroll
    for (int t = 0; t < 4; ++t) {
      const int col = col0 + 16 * t + r;
      const int wch = col / D_;             // 0=Q, 1=K, 2=V  (uniform per wave/tile)
      const int hh  = (col - wch * D_) / HD_;
      const int dd  = col % HD_;
      if (wch == 2) {
        // Vt[b][h][d][token]: 8 consecutive tokens -> one 16B store
        uint4 pay = { pk2(acc[m][t][0], acc[m][t][1]), pk2(acc[m][t][2], acc[m][t][3]),
                      pk2(acc[m][t][4], acc[m][t][5]), pk2(acc[m][t][6], acc[m][t][7]) };
        size_t off = ((size_t)(bb * H_ + hh) * HD_ + dd) * N_ + tok0;
        *(uint4*)(Vt + off) = pay;
      } else {
        const float sc = (wch == 0) ? 0.125f : 1.0f;   // bake hd^-0.5 into Q
        __bf16* dst = (wch == 0 ? Qh : Kh) +
                      ((size_t)(bb * H_ + hh) * N_ + tok0) * HD_ + dd;
#pragma unroll
        for (int j = 0; j < 8; ++j) dst[(size_t)j * HD_] = (__bf16)(acc[m][t][j] * sc);
      }
    }
  }
}

// ---------------- Flash attention (transposed): S^T = K·Q^T, O^T = V^T·P^T ----------------
// One wave handles one (b,h, 16-query tile). Lane = query in all C tiles, so softmax
// stats are per-lane + one shfl_xor(16); P^T feeds the O^T WMMA B-operand in-register.
// V operands are fetched before the softmax VALU block so exp() hides their latency.

__global__ void __launch_bounds__(128) k_attn(const __bf16* __restrict__ Qh,
                                              const __bf16* __restrict__ Kh,
                                              const __bf16* __restrict__ Vt,
                                              __bf16* __restrict__ ao) {
  const int wid  = blockIdx.x * (blockDim.x >> 5) + (threadIdx.x >> 5);
  const int lane = threadIdx.x & 31;
  const int g = lane >> 4, r = lane & 15;
  const int QT = N_ / 16;                  // 64 query tiles per (b,h)
  const int qt = wid % QT;
  const int bh = wid / QT;
  const int bb = bh / H_, hh = bh - bb * H_;
  const int q0 = qt * 16;

  const __bf16* Qb = Qh + ((size_t)bh * N_ + q0 + r) * HD_;   // lane's query column of Q^T
  const __bf16* Kb = Kh + (size_t)bh * N_ * HD_;
  const __bf16* Vb = Vt + (size_t)bh * HD_ * N_;

  // Q^T B-operands (hd=64 -> two K=32 steps), loop-invariant, pre-scaled by hd^-0.5
  v16bf qB[2];
#pragma unroll
  for (int ks = 0; ks < 2; ++ks)
    qB[ks] = load_ab(Qb + 32 * ks + 8 * g, Qb + 32 * ks + 16 + 8 * g);

  float mrun = -3.0e38f, lrun = 0.f;
  v8f o[4];
#pragma unroll
  for (int t = 0; t < 4; ++t) o[t] = zero8();

  for (int kb = 0; kb < N_; kb += 32) {
    // ---- logits: S^T tiles for keys [kb,kb+16) and [kb+16,kb+32) ----
    v8f s0 = zero8(), s1 = zero8();
#pragma unroll
    for (int ks = 0; ks < 2; ++ks) {
      const int kd = 32 * ks;
      const __bf16* k0 = Kb + (size_t)(kb + r) * HD_;
      const __bf16* k1 = Kb + (size_t)(kb + 16 + r) * HD_;
      v16bf a0 = load_ab(k0 + kd + 8 * g, k0 + kd + 16 + 8 * g);
      v16bf a1 = load_ab(k1 + kd + 8 * g, k1 + kd + 16 + 8 * g);
      s0 = WMMA_BF16(a0, qB[ks], s0);
      s1 = WMMA_BF16(a1, qB[ks], s1);
    }
    // ---- issue V-operand loads early; softmax VALU below hides their latency ----
    v16bf va[4];
#pragma unroll
    for (int t = 0; t < 4; ++t) {
      const __bf16* vr = Vb + (size_t)(16 * t + r) * N_;
      va[t] = load_ab(vr + kb + 8 * g, vr + kb + 16 + 8 * g);
    }
    // ---- online softmax: per-lane reduce + one xor-16 (both halves share a query) ----
    float bm = -3.0e38f;
#pragma unroll
    for (int j = 0; j < 8; ++j) bm = fmaxf(bm, fmaxf(s0[j], s1[j]));
    bm = fmaxf(bm, __shfl_xor(bm, 16, 32));
    const float mnew  = fmaxf(mrun, bm);
    const float alpha = __expf(mrun - mnew);
    float ssum = 0.f;
    v16bf p;                                // P^T B-operand: keys kb+8g+i / kb+16+8g+i
#pragma unroll
    for (int j = 0; j < 8; ++j) {
      float e0 = __expf(s0[j] - mnew);
      float e1 = __expf(s1[j] - mnew);
      ssum += e0 + e1;
      p[j]     = (__bf16)e0;
      p[j + 8] = (__bf16)e1;
    }
    ssum += __shfl_xor(ssum, 16, 32);
    lrun = lrun * alpha + ssum;
    mrun = mnew;
    // ---- O^T += V^T(16d x 32keys) · P^T(32keys x 16q), 4 d-tiles ----
#pragma unroll
    for (int t = 0; t < 4; ++t) {
#pragma unroll
      for (int j = 0; j < 8; ++j) o[t][j] *= alpha;
      o[t] = WMMA_BF16(va[t], p, o[t]);
    }
  }

  // normalize + store: lane's query = q0+r, d = 16t+8g+j consecutive -> 16B stores
  const float inv = 1.f / lrun;
#pragma unroll
  for (int t = 0; t < 4; ++t) {
    uint4 pay = { pk2(o[t][0] * inv, o[t][1] * inv), pk2(o[t][2] * inv, o[t][3] * inv),
                  pk2(o[t][4] * inv, o[t][5] * inv), pk2(o[t][6] * inv, o[t][7] * inv) };
    size_t off = ((size_t)bb * N_ + q0 + r) * D_ + hh * HD_ + 16 * t + 8 * g;
    *(uint4*)(ao + off) = pay;
  }
}

// ---------------- Output projection: out = attn @ W_proj + b_proj (fp32 out) ----------------
// Same 32x64 double-buffered tiling as the QKV GEMM.

__global__ void __launch_bounds__(128) k_gemm_proj(const __bf16* __restrict__ ao,
                                                   const __bf16* __restrict__ wT,
                                                   const float* __restrict__ bp,
                                                   float* __restrict__ out) {
  const int wid  = blockIdx.x * (blockDim.x >> 5) + (threadIdx.x >> 5);
  const int lane = threadIdx.x & 31;
  const int g = lane >> 4, r = lane & 15;
  const int NT = D_ / 64;                  // 12 column tiles
  const int mt = wid / NT, nt = wid - mt * NT;
  const int row0 = mt * 32, col0 = nt * 64;

  const __bf16* arow0 = ao + (size_t)(row0 + r) * D_;
  const __bf16* arow1 = ao + (size_t)(row0 + 16 + r) * D_;
  const __bf16* bcol[4];
#pragma unroll
  for (int t = 0; t < 4; ++t) bcol[t] = wT + (size_t)(col0 + 16 * t + r) * D_;

  v8f acc[2][4];
#pragma unroll
  for (int m = 0; m < 2; ++m)
#pragma unroll
    for (int t = 0; t < 4; ++t) acc[m][t] = zero8();

  v16bf Ab[2][2], Bb[2][4];
  auto load_step = [&](int buf, int kb) {
    Ab[buf][0] = load_ab(arow0 + kb + 8 * g, arow0 + kb + 16 + 8 * g);
    Ab[buf][1] = load_ab(arow1 + kb + 8 * g, arow1 + kb + 16 + 8 * g);
#pragma unroll
    for (int t = 0; t < 4; ++t)
      Bb[buf][t] = load_ab(bcol[t] + kb + 8 * g, bcol[t] + kb + 16 + 8 * g);
  };

  load_step(0, 0);
  for (int kb = 0; kb < D_; kb += 32) {
    const int cur = (kb >> 5) & 1;
    if (kb + 32 < D_) load_step(cur ^ 1, kb + 32);
#pragma unroll
    for (int t = 0; t < 4; ++t) {
      acc[0][t] = WMMA_BF16(Ab[cur][0], Bb[cur][t], acc[0][t]);
      acc[1][t] = WMMA_BF16(Ab[cur][1], Bb[cur][t], acc[1][t]);
    }
  }

#pragma unroll
  for (int m = 0; m < 2; ++m) {
#pragma unroll
    for (int t = 0; t < 4; ++t) {
      const int col = col0 + 16 * t + r;
      const float bias = bp[col];
      float* dst = out + (size_t)(row0 + 16 * m + 8 * g) * D_ + col;
#pragma unroll
      for (int j = 0; j < 8; ++j) dst[(size_t)j * D_] = acc[m][t][j] + bias;
    }
  }
}

// ---------------- launch ----------------

extern "C" void kernel_launch(void* const* d_in, const int* in_sizes, int n_in,
                              void* d_out, int out_size, void* d_ws, size_t ws_size,
                              hipStream_t stream) {
  (void)in_sizes; (void)n_in; (void)out_size; (void)ws_size;
  const float* x  = (const float*)d_in[0];   // [B,N,D]
  const float* wq = (const float*)d_in[1];   // [D,3D]
  const float* wp = (const float*)d_in[2];   // [D,D]
  const float* bp = (const float*)d_in[3];   // [D]
  float* out = (float*)d_out;                // [B,N,D]

  char* ws = (char*)d_ws;
  size_t off = 0;
  __bf16* xb  = (__bf16*)(ws + off); off += (size_t)BN_ * D_ * 2;   // x bf16
  __bf16* wqT = (__bf16*)(ws + off); off += (size_t)TD_ * D_ * 2;   // W_qkv^T bf16
  __bf16* wpT = (__bf16*)(ws + off); off += (size_t)D_  * D_ * 2;   // W_proj^T bf16
  __bf16* Qh  = (__bf16*)(ws + off); off += (size_t)BN_ * D_ * 2;   // [B,H,N,hd] (pre-scaled)
  __bf16* Kh  = (__bf16*)(ws + off); off += (size_t)BN_ * D_ * 2;   // [B,H,N,hd]
  __bf16* Vt  = (__bf16*)(ws + off); off += (size_t)BN_ * D_ * 2;   // [B,H,hd,N]
  __bf16* ao  = (__bf16*)(ws + off); off += (size_t)BN_ * D_ * 2;   // attn out [B,N,D]

  // 1) fp32 -> bf16 (x; transposed weights for contiguous WMMA B-operand rows)
  {
    int n4 = BN_ * D_ / 4;
    k_cvt_x<<<(n4 + 255) / 256, 256, 0, stream>>>(x, xb, n4);
    int nq = D_ * TD_;
    k_cvt_wT<<<(nq + 255) / 256, 256, 0, stream>>>(wq, wqT, D_, TD_);
    int np = D_ * D_;
    k_cvt_wT<<<(np + 255) / 256, 256, 0, stream>>>(wp, wpT, D_, D_);
  }
  // 2) QKV GEMM (32x64 tiles), scatter into Q/K/Vt head layouts
  {
    int waves = (BN_ / 32) * (TD_ / 64);   // 18432
    k_gemm_qkv<<<waves / 4, 128, 0, stream>>>(xb, wqT, Qh, Kh, Vt);
  }
  // 3) flash attention (transposed), one wave per 16-query tile
  {
    int waves = B_ * H_ * (N_ / 16);       // 12288
    k_attn<<<waves / 4, 128, 0, stream>>>(Qh, Kh, Vt, ao);
  }
  // 4) output projection + bias (32x64 tiles)
  {
    int waves = (BN_ / 32) * (D_ / 64);    // 6144
    k_gemm_proj<<<waves / 4, 128, 0, stream>>>(ao, wpT, bp, out);
  }
}
